// GCNGRU_22299470201220
// MI455X (gfx1250) — matrix operand order repaired and checked by reading, not compiled, not run on hardware
//
#include <hip/hip_runtime.h>

#define NN    20000
#define NHID  256
#define NOUT  128
#define HDIM  128
#define TT    16
#define EE    320000
#define EPP   100000
#define NHIDE 256

typedef __attribute__((ext_vector_type(16))) _Float16     v16h;
typedef __attribute__((ext_vector_type(8)))  float        v8f;
typedef __attribute__((ext_vector_type(4)))  unsigned int u32x4;
typedef __attribute__((ext_vector_type(8)))  int          i32x8;
typedef __attribute__((ext_vector_type(4)))  int          i32x4;

static __device__ __forceinline__ v8f wmma16(v16h a, v16h b, v8f c) {
  // emits v_wmma_f32_16x16x32_f16
  return __builtin_amdgcn_wmma_f32_16x16x32_f16(false, a, false, b, (short)0, c, false, false);
}

// ---- Tensor Data Mover: 2D tile (rows x cols f32, row stride = cols) -------
// D# per CDNA5 ISA; 6-arg builtin (clang-23 form).
static __device__ __forceinline__ unsigned lds_off_of(const void* p) {
  // flat LDS addresses carry the LDS byte offset in addr[31:0] (ISA §10.2)
  return (unsigned)(size_t)p;
}

static __device__ __forceinline__ void tdm_load_tile_f32(const float* gsrc,
                                                         unsigned lds_addr,
                                                         int rows, int cols,
                                                         int tensor_rows) {
  unsigned long long ga = (unsigned long long)(size_t)gsrc;
  u32x4 g0;
  g0[0] = 1u;                                           // count=1, no gather
  g0[1] = lds_addr;                                     // lds_addr [63:32]
  g0[2] = (unsigned)(ga & 0xffffffffu);                 // global_addr[31:0]
  g0[3] = (unsigned)((ga >> 32) & 0x1ffffffu)           // global_addr[56:32]
        | (2u << 30);                                   // type=2 ("image")
  i32x8 g1;
  g1[0] = 2 << 16;                                      // data_size = 4 bytes
  g1[1] = (cols & 0xffff) << 16;                        // tensor_dim0[15:0]
  g1[2] = ((unsigned)cols >> 16)
        | ((tensor_rows & 0xffff) << 16);               // dim0 hi | tensor_dim1 lo
  g1[3] = (((unsigned)tensor_rows >> 16) & 0xffff)
        | ((cols & 0xffff) << 16);                      // tensor_dim1 hi | tile_dim0
  g1[4] = rows & 0xffff;                                // tile_dim1 (tile_dim2 = 0)
  g1[5] = cols;                                         // tensor_dim0_stride[31:0]
  g1[6] = 0;
  g1[7] = 0;
  i32x4 z4 = {};
  i32x8 z8 = {};
  // groups 2/3 unused (<=2D tensor); extra descriptor words zero
  __builtin_amdgcn_tensor_load_to_lds(g0, g1, z4, z4, z8, 0);
}

// ---- WMMA fragment helpers (ISA 7.12.2 layouts, wave32) -------------------

// A: 16x32 f16 from row-major f32; lane r=lane&15 holds row r.
// element i -> K = (i&7) + 8*half + 16*(i>>3)  (two contiguous 8-float runs)
static __device__ __forceinline__ v16h load_a_g(const float* src, int ld, int row0,
                                                int k0, int lane, bool relu) {
  int r = lane & 15, hf = lane >> 4;
  const float* p = src + (size_t)(row0 + r) * ld + k0;
  v16h a;
#pragma unroll
  for (int i = 0; i < 16; ++i) {
    int K = (i & 7) + 8 * hf + ((i >> 3) << 4);
    float v = p[K];
    if (relu) v = v > 0.f ? v : 0.f;
    a[i] = (_Float16)v;
  }
  return a;
}

// A from LDS f32 tile (row-major, stride ld)
static __device__ __forceinline__ v16h load_a_lds_f32(const float* src, int ld,
                                                      int k0, int lane) {
  int r = lane & 15, hf = lane >> 4;
  const float* p = src + r * ld + k0;
  v16h a;
#pragma unroll
  for (int i = 0; i < 16; ++i)
    a[i] = (_Float16)p[(i & 7) + 8 * hf + ((i >> 3) << 4)];
  return a;
}

// A from LDS f16 tile (row-major, stride ld)
static __device__ __forceinline__ v16h load_a_lds(const _Float16* src, int ld,
                                                  int k0, int lane) {
  int r = lane & 15, hf = lane >> 4;
  const _Float16* p = src + r * ld + k0;
  v16h a;
#pragma unroll
  for (int i = 0; i < 16; ++i) a[i] = p[(i & 7) + 8 * hf + ((i >> 3) << 4)];
  return a;
}

// Pre-packed B fragment: buf[frag*32 + lane] = 16 contiguous halves per lane
static __device__ __forceinline__ v16h load_b_p(const v16h* __restrict__ buf,
                                                int frag, int lane) {
  return buf[(size_t)frag * 32 + lane];
}

// C/D: element j -> row = j + 8*half, col = lane&15
static __device__ __forceinline__ void store_c_g(float* dst, int ld, int row0,
                                                 int n0, int lane, v8f c) {
  int n = lane & 15, hf = lane >> 4;
#pragma unroll
  for (int j = 0; j < 8; ++j)
    dst[(size_t)(row0 + j + 8 * hf) * ld + n0 + n] = c[j];
}

// ---- weight packing kernels -----------------------------------------------
// dst layout: frag f = nt*Ktiles + kt; element idx = (f*32 + lane)*16 + i
// B element (K = kt*32 + (i&7) + 8*hf + 16*(i>>3), n = nt*16 + (lane&15))

__global__ void pack_b_kernel(const float* __restrict__ src, int ld, int Ktiles,
                              _Float16* __restrict__ dst, int total) {
  int idx = blockIdx.x * blockDim.x + threadIdx.x;
  if (idx >= total) return;
  int i = idx & 15, lane = (idx >> 4) & 31, f = idx >> 9;
  int kt = f % Ktiles, nt = f / Ktiles;
  int K = kt * 32 + (i & 7) + 8 * (lane >> 4) + ((i >> 3) << 4);
  int n = nt * 16 + (lane & 15);
  dst[idx] = (_Float16)src[(size_t)K * ld + n];   // row-major [K, Ncols]
}

__global__ void pack_bT_kernel(const float* __restrict__ W, int ldw, int Ktiles,
                               _Float16* __restrict__ dst, int total) {
  int idx = blockIdx.x * blockDim.x + threadIdx.x;
  if (idx >= total) return;
  int i = idx & 15, lane = (idx >> 4) & 31, f = idx >> 9;
  int kt = f % Ktiles, nt = f / Ktiles;
  int K = kt * 32 + (i & 7) + 8 * (lane >> 4) + ((i >> 3) << 4);
  int n = nt * 16 + (lane & 15);
  dst[idx] = (_Float16)W[(size_t)n * ldw + K];    // B = W^T, W row-major [Ncols, K]
}

// ---- helper kernels --------------------------------------------------------

__global__ void fill_zero_kernel(float* dst, int total) {
  int i = blockIdx.x * blockDim.x + threadIdx.x;
  if (i < total) dst[i] = 0.f;
}

__global__ void broadcast_bias_kernel(float* dst, const float* bias, int total, int colmask) {
  int i = blockIdx.x * blockDim.x + threadIdx.x;
  if (i < total) dst[i] = bias[i & colmask];
}

// SpMM: out[row] += val * B[col]  (out pre-initialized with bias)
__global__ void spmm_atomic_kernel(const int* __restrict__ rows,
                                   const int* __restrict__ cols,
                                   const float* __restrict__ vals,
                                   const float* __restrict__ B,
                                   float* __restrict__ outb, int ncols) {
  int per = ncols >> 5;
  int tid = blockIdx.x * blockDim.x + threadIdx.x;
  if (tid >= EE * per) return;
  int e = tid / per, part = tid - e * per;
  int r = rows[e], c = cols[e];
  float v = vals[e];
  const float4* src = (const float4*)(B + (size_t)c * ncols + part * 32);
  float* dst = outb + (size_t)r * ncols + part * 32;
#pragma unroll
  for (int q = 0; q < 8; ++q) {
    float4 w = src[q];
    atomicAdd(dst + q * 4 + 0, v * w.x);
    atomicAdd(dst + q * 4 + 1, v * w.y);
    atomicAdd(dst + q * 4 + 2, v * w.z);
    atomicAdd(dst + q * 4 + 3, v * w.w);
  }
}

// y = relu(x) @ W1   x:[N,256] W1p packed [Ktiles=8, Ntiles=8]
// one wave = 16 rows x all 128 cols; A fragment reused across 8 N-tiles
__global__ void gcn_dense_kernel(const float* __restrict__ x,
                                 const _Float16* __restrict__ W1p,
                                 float* __restrict__ y) {
  int lane = threadIdx.x & 31;
  int wave = threadIdx.x >> 5;
  int m0 = (blockIdx.x * 2 + wave) * 16;
  const v16h* Bp = (const v16h*)W1p;
  v8f acc[8] = {};
#pragma unroll
  for (int kt = 0; kt < 8; ++kt) {
    v16h a = load_a_g(x, NHID, m0, kt * 32, lane, true);
#pragma unroll
    for (int nt = 0; nt < 8; ++nt)
      acc[nt] = wmma16(a, load_b_p(Bp, nt * 8 + kt, lane), acc[nt]);
  }
#pragma unroll
  for (int nt = 0; nt < 8; ++nt)
    store_c_g(y, NOUT, m0, nt * 16, lane, acc[nt]);
}

// Fused GRU step: one wave owns 16 nodes. x/h tiles staged to LDS by the
// Tensor Data Mover (async DMA, waited via s_wait_tensorcnt); A fragments
// register-cached; gi/gh kept separate in LDS for the n-gate.
__global__ void gru_step_kernel(const float* __restrict__ xin,     // [N,128]
                                float* __restrict__ h,             // [N,128] in/out
                                const _Float16* __restrict__ Wihp, // packed [24,4] frags
                                const _Float16* __restrict__ Whhp,
                                const float* __restrict__ b_ih,
                                const float* __restrict__ b_hh) {
  __shared__ float xtile[16 * HDIM];
  __shared__ float htile[16 * HDIM];
  __shared__ float gi[16 * 384];
  __shared__ float gh[16 * 384];
  int lane = threadIdx.x;
  int m0 = blockIdx.x * 16;
  const v16h* Bi = (const v16h*)Wihp;
  const v16h* Bh = (const v16h*)Whhp;

  // async DMA both tiles into LDS (TENSORcnt tracked)
  tdm_load_tile_f32(xin + (size_t)m0 * HDIM, lds_off_of(xtile), 16, HDIM, NN);
  tdm_load_tile_f32(h   + (size_t)m0 * HDIM, lds_off_of(htile), 16, HDIM, NN);
  __builtin_amdgcn_s_wait_tensorcnt(0);

  v16h ax[4], ahh[4];
#pragma unroll
  for (int kt = 0; kt < 4; ++kt) {
    ax[kt]  = load_a_lds_f32(xtile, HDIM, kt * 32, lane);
    ahh[kt] = load_a_lds_f32(htile, HDIM, kt * 32, lane);
  }

  for (int ct = 0; ct < 24; ++ct) {
    int n0 = ct * 16;
    float bi = b_ih[n0 + (lane & 15)];
    float bv = b_hh[n0 + (lane & 15)];
    v8f ai, av;
#pragma unroll
    for (int j = 0; j < 8; ++j) { ai[j] = bi; av[j] = bv; }
#pragma unroll
    for (int kt = 0; kt < 4; ++kt) {
      ai = wmma16(ax[kt],  load_b_p(Bi, ct * 4 + kt, lane), ai);
      av = wmma16(ahh[kt], load_b_p(Bh, ct * 4 + kt, lane), av);
    }
    int c = lane & 15, hf = lane >> 4;
#pragma unroll
    for (int j = 0; j < 8; ++j) {
      gi[(j + 8 * hf) * 384 + n0 + c] = ai[j];
      gh[(j + 8 * hf) * 384 + n0 + c] = av[j];
    }
  }
  __syncthreads();
  for (int idx = lane; idx < 16 * HDIM; idx += 32) {
    int node = idx >> 7, c = idx & (HDIM - 1);
    float ir = gi[node * 384 + c],        hr = gh[node * 384 + c];
    float iz = gi[node * 384 + 128 + c],  hz = gh[node * 384 + 128 + c];
    float in_ = gi[node * 384 + 256 + c], hn = gh[node * 384 + 256 + c];
    float r = 1.f / (1.f + expf(-(ir + hr)));
    float z = 1.f / (1.f + expf(-(iz + hz)));
    float nn = tanhf(in_ + r * hn);
    float hold = htile[node * HDIM + c];           // reuse DMA'd h tile
    h[(size_t)(m0 + node) * HDIM + c] = (1.f - z) * nn + z * hold;
  }
}

__global__ void bn_stats_kernel(const float* __restrict__ hbuf,
                                float* __restrict__ mean, float* __restrict__ rstd) {
  __shared__ float ss[256], sq[256];
  int c = blockIdx.x, t = threadIdx.x;
  float s = 0.f, q = 0.f;
  for (int n = t; n < NN; n += 256) {
    float v = hbuf[(size_t)n * HDIM + c];
    s += v; q += v * v;
  }
  ss[t] = s; sq[t] = q;
  __syncthreads();
  for (int off = 128; off; off >>= 1) {
    if (t < off) { ss[t] += ss[t + off]; sq[t] += sq[t + off]; }
    __syncthreads();
  }
  if (t == 0) {
    float m = ss[0] / (float)NN;
    float var = sq[0] / (float)NN - m * m;
    mean[c] = m;
    rstd[c] = rsqrtf(var + 1e-5f);
  }
}

__global__ void bn_apply_kernel(const float* __restrict__ hbuf,
                                const float* __restrict__ mean,
                                const float* __restrict__ rstd,
                                const float* __restrict__ gamma,
                                const float* __restrict__ beta,
                                float* __restrict__ femb) {
  int i = blockIdx.x * blockDim.x + threadIdx.x;
  if (i < NN * HDIM) {
    int c = i & (HDIM - 1);
    femb[i] = (hbuf[i] - mean[c]) * rstd[c] * gamma[c] + beta[c];
  }
}

// Fused edge predictor: gather -> WMMA MLP -> log_softmax; one wave = 16 edges
__global__ void edge_mlp_kernel(const float* __restrict__ femb,
                                const int* __restrict__ edges,     // [2, EP]
                                const _Float16* __restrict__ We1p, // packed
                                const float* __restrict__ be1,
                                const float* __restrict__ We2,     // [256,2]
                                const float* __restrict__ be2,
                                float* __restrict__ out) {
  __shared__ _Float16 hcat[16 * 256];
  __shared__ _Float16 hmid[16 * 256];
  int lane = threadIdx.x;
  int e0 = blockIdx.x * 16;
  const v16h* Bp = (const v16h*)We1p;

  for (int idx = lane; idx < 16 * 256; idx += 32) {
    int e = idx >> 8, c = idx & 255;
    int node = (c < 128) ? edges[e0 + e] : edges[EPP + e0 + e];
    hcat[idx] = (_Float16)femb[(size_t)node * HDIM + (c & 127)];
  }
  __syncthreads();

  v16h afr[8];
#pragma unroll
  for (int kt = 0; kt < 8; ++kt) afr[kt] = load_a_lds(hcat, 256, kt * 32, lane);

  for (int nt = 0; nt < 16; ++nt) {
    float bv = be1[nt * 16 + (lane & 15)];
    v8f acc;
#pragma unroll
    for (int j = 0; j < 8; ++j) acc[j] = bv;
#pragma unroll
    for (int kt = 0; kt < 8; ++kt)
      acc = wmma16(afr[kt], load_b_p(Bp, nt * 8 + kt, lane), acc);
    int c = lane & 15, hf = lane >> 4;
#pragma unroll
    for (int j = 0; j < 8; ++j) {
      float v = acc[j];
      hmid[(j + 8 * hf) * 256 + nt * 16 + c] = (_Float16)(v > 0.f ? v : 0.f);
    }
  }
  __syncthreads();

  int e = lane >> 1, cls = lane & 1;
  float acc = be2[cls];
  for (int k = 0; k < 256; ++k)
    acc += (float)hmid[e * 256 + k] * We2[k * 2 + cls];
  float other = __shfl_xor(acc, 1);
  float m = fmaxf(acc, other);
  float lse = m + logf(expf(acc - m) + expf(other - m));
  out[(size_t)(e0 + e) * 2 + cls] = acc - lse;
}

// ---- orchestration ---------------------------------------------------------

extern "C" void kernel_launch(void* const* d_in, const int* in_sizes, int n_in,
                              void* d_out, int out_size, void* d_ws, size_t ws_size,
                              hipStream_t stream) {
  const int*   adj_rows = (const int*)d_in[0];    // [T,E]
  const int*   adj_cols = (const int*)d_in[1];
  const float* adj_vals = (const float*)d_in[2];
  const int*   edges    = (const int*)d_in[3];    // [2,EP]
  const float* W0   = (const float*)d_in[4];
  const float* b0   = (const float*)d_in[5];
  const float* W1   = (const float*)d_in[6];
  const float* b1   = (const float*)d_in[7];
  const float* W_ih = (const float*)d_in[8];
  const float* W_hh = (const float*)d_in[9];
  const float* b_ih = (const float*)d_in[10];
  const float* b_hh = (const float*)d_in[11];
  const float* gamma = (const float*)d_in[12];
  const float* beta  = (const float*)d_in[13];
  const float* We1 = (const float*)d_in[14];
  const float* be1 = (const float*)d_in[15];
  const float* We2 = (const float*)d_in[16];
  const float* be2 = (const float*)d_in[17];
  float* out = (float*)d_out;

  // workspace layout: f32 region (~61.5 MB) then packed f16 weights (~400 KB)
  float* ws   = (float*)d_ws;
  float* x    = ws;                          // [N, 256]
  float* y    = x    + (size_t)NN * NHID;    // [N, 128]
  float* outt = y    + (size_t)NN * NOUT;    // [N, 128]
  float* h    = outt + (size_t)NN * NOUT;    // [N, 128]
  float* femb = h    + (size_t)NN * HDIM;    // [N, 128]
  float* mean = femb + (size_t)NN * HDIM;    // [128]
  float* rstd = mean + HDIM;                 // [128]
  _Float16* W1p  = (_Float16*)(rstd + HDIM); // 256x128 packed
  _Float16* Wihp = W1p  + 32768;             // 128x384 packed (B = W_ih^T)
  _Float16* Whhp = Wihp + 49152;             // 128x384 packed (B = W_hh^T)
  _Float16* We1p = Whhp + 49152;             // 256x256 packed

  const int NX = NN * NHID;   // 5.12M
  const int NO = NN * NOUT;   // 2.56M

  // pack weights into WMMA B-fragment layout (contiguous b128 loads)
  pack_b_kernel <<<32768 / 256, 256, 0, stream>>>(W1,  NOUT, 8, W1p,  32768);
  pack_bT_kernel<<<49152 / 256, 256, 0, stream>>>(W_ih, HDIM, 4, Wihp, 49152);
  pack_bT_kernel<<<49152 / 256, 256, 0, stream>>>(W_hh, HDIM, 4, Whhp, 49152);
  pack_b_kernel <<<65536 / 256, 256, 0, stream>>>(We1, NHIDE, 8, We1p, 65536);

  fill_zero_kernel<<<(NO + 255) / 256, 256, 0, stream>>>(h, NO);

  for (int t = 0; t < TT; ++t) {
    const int*   rt = adj_rows + (size_t)t * EE;
    const int*   ct = adj_cols + (size_t)t * EE;
    const float* vt = adj_vals + (size_t)t * EE;

    // layer 0: x = b0 + A @ W0   (relu fused into next GEMM's A-load)
    broadcast_bias_kernel<<<(NX + 255) / 256, 256, 0, stream>>>(x, b0, NX, NHID - 1);
    spmm_atomic_kernel<<<(EE * 8 + 255) / 256, 256, 0, stream>>>(rt, ct, vt, W0, x, NHID);

    // y = relu(x) @ W1
    gcn_dense_kernel<<<NN / 32, 64, 0, stream>>>(x, W1p, y);

    // layer 1: out_t = b1 + A @ y
    broadcast_bias_kernel<<<(NO + 255) / 256, 256, 0, stream>>>(outt, b1, NO, NOUT - 1);
    spmm_atomic_kernel<<<(EE * 4 + 255) / 256, 256, 0, stream>>>(rt, ct, vt, y, outt, NOUT);

    // GRU cell (TDM-staged tiles, in-place h update)
    gru_step_kernel<<<NN / 16, 32, 0, stream>>>(outt, h, Wihp, Whhp, b_ih, b_hh);
  }

  bn_stats_kernel<<<HDIM, 256, 0, stream>>>(h, mean, rstd);
  bn_apply_kernel<<<(NO + 255) / 256, 256, 0, stream>>>(h, mean, rstd, gamma, beta, femb);

  edge_mlp_kernel<<<EPP / 16, 32, 0, stream>>>(femb, edges, We1p, be1, We2, be2, out);
}
